// CRF_20358144983237
// MI455X (gfx1250) — compile-verified, matrix operand order checked by reading, not run on hardware
//
#include <hip/hip_runtime.h>
#include <math.h>

// ---------------- problem constants ----------------
#define HH 96
#define WW 96
#define NPIX (HH * WW)   // 9216
#define NC 21            // classes
#define CP 32            // classes padded to 2 WMMA row tiles
#define NF 8             // feature dims padded (5 real)
#define RAD 35           // gaussian radius (sig_sq=36 even -> r=35)
#define KS 71            // gaussian taps
#define COMPAT_BF 4.0f
#define COMPAT_SP 2.0f

typedef __attribute__((ext_vector_type(16))) __bf16        v16bf;
typedef __attribute__((ext_vector_type(8)))  float         v8f;
typedef __attribute__((ext_vector_type(2)))  float         v2f;
typedef __attribute__((ext_vector_type(8)))  unsigned int  v8u;
typedef __attribute__((ext_vector_type(4)))  int           v4i;

// CDNA5 async global->LDS path (guarded: falls back to VGPR staging if the
// toolchain doesn't declare the builtins)
#if defined(__HIP_DEVICE_COMPILE__) && \
    __has_builtin(__builtin_amdgcn_global_load_async_to_lds_b128) && \
    __has_builtin(__builtin_amdgcn_s_wait_asynccnt)
#define USE_ASYNC_LDS 1
#else
#define USE_ASYNC_LDS 0
#endif

#if USE_ASYNC_LDS
typedef __attribute__((address_space(1))) v4i g_v4i;
typedef __attribute__((address_space(3))) v4i l_v4i;
__device__ __forceinline__ void async_cp16(const void* g, const void* l) {
  // generic LDS address -> LDS offset via low-32-bit truncation (flat aperture rule)
  __builtin_amdgcn_global_load_async_to_lds_b128(
      (g_v4i*)(unsigned long long)(size_t)g,
      (l_v4i*)(unsigned int)(size_t)l, 0, 0);
}
#endif

__device__ __forceinline__ unsigned short f2bf_bits(float f) {
  unsigned int u = __float_as_uint(f);
  u += 0x7FFFu + ((u >> 16) & 1u);            // round-to-nearest-even
  return (unsigned short)(u >> 16);
}
// pack two floats into one dword of bf16 (lo, hi)
__device__ __forceinline__ unsigned int pack2bf(float lo, float hi) {
  unsigned int ul = __float_as_uint(lo); ul += 0x7FFFu + ((ul >> 16) & 1u);
  unsigned int uh = __float_as_uint(hi); uh += 0x7FFFu + ((uh >> 16) & 1u);
  return (ul >> 16) | (uh & 0xFFFF0000u);
}

// ---------------- init: scaled features + |f|^2 ----------------
__global__ void k_init_feat(const float* __restrict__ ref,
                            const float* __restrict__ kstd,
                            float* __restrict__ F, float* __restrict__ sq) {
  int p = blockIdx.x * blockDim.x + threadIdx.x;
  if (p >= NPIX) return;
  int y = p / WW, x = p % WW;
  float f[NF];
  f[0] = (float)y / kstd[0];
  f[1] = (float)x / kstd[1];
  f[2] = ref[0 * NPIX + p] / kstd[2];
  f[3] = ref[1 * NPIX + p] / kstd[3];
  f[4] = ref[2 * NPIX + p] / kstd[4];
  f[5] = 0.f; f[6] = 0.f; f[7] = 0.f;
  float s = 0.f;
#pragma unroll
  for (int d = 0; d < NF; ++d) { F[p * NF + d] = f[d]; s += f[d] * f[d]; }
  sq[p] = s;
}

// ---------------- init: U = log(clip(u)), q = softmax(U) ----------------
__global__ void k_init_uq(const float* __restrict__ unary,
                          float* __restrict__ U, float* __restrict__ q) {
  int p = blockIdx.x * blockDim.x + threadIdx.x;
  if (p >= NPIX) return;
  float u[NC]; float mx = -1e30f;
#pragma unroll
  for (int c = 0; c < NC; ++c) {
    float v = unary[c * NPIX + p];
    v = fminf(fmaxf(v, 1e-5f), 1.0f);
    v = __logf(v);
    U[c * NPIX + p] = v;
    u[c] = v;
    mx = fmaxf(mx, v);
  }
  float s = 0.f;
#pragma unroll
  for (int c = 0; c < NC; ++c) { u[c] = __expf(u[c] - mx); s += u[c]; }
  float inv = __frcp_rn(s);
#pragma unroll
  for (int c = 0; c < NC; ++c) q[c * NPIX + p] = u[c] * inv;
}

// ---------------- 1D separable gaussian taps ----------------
__global__ void k_g1d(float* __restrict__ g) {
  __shared__ float w[KS];
  __shared__ float s;
  int t = threadIdx.x;
  if (t < KS) { float d = (float)(t - RAD); w[t] = __expf(-(d * d) / 72.0f); }
  __syncthreads();
  if (t == 0) { float a = 0.f; for (int i = 0; i < KS; ++i) a += w[i]; s = a; }
  __syncthreads();
  if (t < KS) g[t] = w[t] * __frcp_rn(s);
}

// ---------------- norm_j = sqrt(sum_i K[i][j]) + 1e-8 ----------------
__global__ void __launch_bounds__(256) k_norm(const float* __restrict__ F,
                                              const float* __restrict__ sq,
                                              float* __restrict__ nrm) {
  __shared__ float sF[256 * NF];
  __shared__ float sS[256];
  int tid = threadIdx.x;
  int j = blockIdx.x * 256 + tid;
  float f0 = F[j * NF + 0], f1 = F[j * NF + 1], f2 = F[j * NF + 2],
        f3 = F[j * NF + 3], f4 = F[j * NF + 4];
  float sj = sq[j];
  float acc = 0.f;
  for (int i0 = 0; i0 < NPIX; i0 += 256) {
    if (i0 + 256 < NPIX)
      __builtin_prefetch(&F[(i0 + 256) * NF + tid], 0, 0);  // global_prefetch_b8
#pragma unroll
    for (int d = 0; d < NF; ++d) sF[tid * NF + d] = F[(i0 + tid) * NF + d];
    sS[tid] = sq[i0 + tid];
    __syncthreads();
    for (int i = 0; i < 256; ++i) {
      float dot = f0 * sF[i * NF + 0] + f1 * sF[i * NF + 1] + f2 * sF[i * NF + 2]
                + f3 * sF[i * NF + 3] + f4 * sF[i * NF + 4];
      float d2 = sj + sS[i] - 2.f * dot;
      acc += __expf(-0.5f * fmaxf(d2, 0.f));
    }
    __syncthreads();
  }
  nrm[j] = __fsqrt_rn(acc) + 1e-8f;
}

// -------- prep: bf16(q/norm), pixel-pair interleaved, dword writes --------
// dword (p>>1)*32 + c holds (pixel even, c) lo and (pixel odd, c) hi.
__global__ void k_prep(const float* __restrict__ q, const float* __restrict__ nrm,
                       unsigned int* __restrict__ qn32) {
  int pp = blockIdx.x * blockDim.x + threadIdx.x;   // pixel pair
  if (pp >= NPIX / 2) return;
  int pe = 2 * pp, po = pe + 1;
  float inve = __frcp_rn(nrm[pe]);
  float invo = __frcp_rn(nrm[po]);
  unsigned int* dst = qn32 + (size_t)pp * CP;
#pragma unroll
  for (int c = 0; c < CP; ++c) {
    float ve = (c < NC) ? q[c * NPIX + pe] * inve : 0.f;
    float vo = (c < NC) ? q[c * NPIX + po] * invo : 0.f;
    dst[c] = pack2bf(ve, vo);
  }
}

// ------------- fused bilateral: out = qn @ K ------------------------------
// M = 32 padded classes (2 row tiles), N = 16 out pixels per wave,
// inner = 9216 pixels in chunks of 32.
// Gram tiles G[i][j] = F_i . F_j via V_WMMA_F32_16X16X4_F32 (K=4, x2 for 8 dims),
// then d2/exp in-lane -> bf16 B operand; qn A-fragments loaded as dwords.
__global__ void __launch_bounds__(256) k_bilateral(
    const float* __restrict__ F, const float* __restrict__ sq,
    const unsigned int* __restrict__ qn32,   // interleaved bf16 pairs
    const float* __restrict__ nrm,
    float* __restrict__ qbf)                 // [NPIX][CP] f32
{
  __shared__ float        sF[2][32 * NF];    // inner-pixel features
  __shared__ float        sS[2][32];         // inner |f|^2
  __shared__ unsigned int sQu[2][512];       // 32x32 bf16 qn chunk (dwords)

  const int tid  = threadIdx.x;
  const int wave = tid >> 5;
  const int lane = tid & 31;
  const int n    = lane & 15;      // out-pixel column within wave tile
  const int half = lane >> 4;
  const int jBase = blockIdx.x * 128 + wave * 16;
  const int j = jBase + n;         // this lane's output pixel

  // loop-invariant Gram B operand: features of out pixel j
  // f32 B 4x16 layout: VGPR0 = K0(lanes0-15)/K2(lanes16-31), VGPR1 = K1/K3
  const v2f bg0 = { F[j * NF + (half ? 2 : 0)], F[j * NF + (half ? 3 : 1)] };
  const v2f bg1 = { half ? 0.f : F[j * NF + 4], 0.f };
  const float sqj = sq[j];

  v8f acc0 = {}, acc1 = {};

  auto stage = [&](int nb, int nx) {
#if USE_ASYNC_LDS
    // 200 x b128 async DMA: 64 (features) + 128 (qn chunk) + 8 (|f|^2)
    if (tid < 64) {
      async_cp16(&F[nx * NF + tid * 4], &sF[nb][tid * 4]);
    } else if (tid < 192) {
      int u = tid - 64;
      async_cp16(&qn32[nx * 16 + u * 4], &sQu[nb][u * 4]);
    } else if (tid < 200) {
      int u = tid - 192;
      async_cp16(&sq[nx + u * 4], &sS[nb][u * 4]);
    }
#else
    sF[nb][tid] = F[nx * NF + tid];
    if (tid < 32) sS[nb][tid] = sq[nx + tid];
    sQu[nb][tid]       = qn32[nx * 16 + tid];
    sQu[nb][tid + 256] = qn32[nx * 16 + tid + 256];
#endif
  };

  stage(0, 0);                               // preload chunk 0

  for (int i0 = 0; i0 < NPIX; i0 += 32) {
    const int buf = (i0 >> 5) & 1;
#if USE_ASYNC_LDS
    __builtin_amdgcn_s_wait_asynccnt(0);     // own async loads landed
#endif
    __syncthreads();                         // everyone's staging visible
    if (i0 + 32 < NPIX) stage(buf ^ 1, i0 + 32);

    // ---- Gram tiles: G0 = inner rows 0-15, G1 = inner rows 16-31 ----
    const int il0 = n;            // f32 A 16x4 layout: lane row = lane%16
    const int il1 = n + 16;
    const int d0 = half ? 2 : 0, d1 = half ? 3 : 1;
    v2f a0 = { sF[buf][il0 * NF + d0], sF[buf][il0 * NF + d1] };
    v2f a1 = { half ? 0.f : sF[buf][il0 * NF + 4], 0.f };
    v8f G0 = {};
    G0 = __builtin_amdgcn_wmma_f32_16x16x4_f32(false, a0, false, bg0, (short)0, G0, false, false);
    G0 = __builtin_amdgcn_wmma_f32_16x16x4_f32(false, a1, false, bg1, (short)0, G0, false, false);
    v2f a2 = { sF[buf][il1 * NF + d0], sF[buf][il1 * NF + d1] };
    v2f a3 = { half ? 0.f : sF[buf][il1 * NF + 4], 0.f };
    v8f G1 = {};
    G1 = __builtin_amdgcn_wmma_f32_16x16x4_f32(false, a2, false, bg0, (short)0, G1, false, false);
    G1 = __builtin_amdgcn_wmma_f32_16x16x4_f32(false, a3, false, bg1, (short)0, G1, false, false);

    // ---- d2 -> exp -> bf16 B fragment (layout matches C/D of Gram) ----
    float e0[8], e1[8];
#pragma unroll
    for (int r = 0; r < 8; ++r) {
      float d2a = sS[buf][r + half * 8]      + sqj - 2.f * G0[r];
      float d2b = sS[buf][16 + r + half * 8] + sqj - 2.f * G1[r];
      e0[r] = __expf(-0.5f * fmaxf(d2a, 0.f));
      e1[r] = __expf(-0.5f * fmaxf(d2b, 0.f));
    }
    v8u bu;
#pragma unroll
    for (int v = 0; v < 4; ++v) {
      bu[v]     = pack2bf(e0[2 * v], e0[2 * v + 1]);
      bu[v + 4] = pack2bf(e1[2 * v], e1[2 * v + 1]);
    }
    const v16bf b = __builtin_bit_cast(v16bf, bu);

    // ---- qn A fragments (classes 0-15 and 16-31), dword loads from LDS ----
    v8u au0, au1;
#pragma unroll
    for (int v = 0; v < 8; ++v) {
      const int kh = ((v < 4) ? v : 4 + v) + half * 4;   // pixel-pair index
      au0[v] = sQu[buf][kh * 32 + n];
      au1[v] = sQu[buf][kh * 32 + 16 + n];
    }
    const v16bf aA = __builtin_bit_cast(v16bf, au0);
    const v16bf aB = __builtin_bit_cast(v16bf, au1);

    acc0 = __builtin_amdgcn_wmma_f32_16x16x32_bf16(false, aA, false, b,
                                                   (short)0, acc0, false, false);
    acc1 = __builtin_amdgcn_wmma_f32_16x16x32_bf16(false, aB, false, b,
                                                   (short)0, acc1, false, false);
  }

  // D layout: VGPR r -> class r (lanes 0-15) / r+8 (lanes 16-31), pixel = lane%16.
  // acc[r] maps to contiguous classes -> two 32-byte vector stores.
  const float invn = __frcp_rn(nrm[j]);
  v8f o0, o1;
#pragma unroll
  for (int r = 0; r < 8; ++r) { o0[r] = acc0[r] * invn; o1[r] = acc1[r] * invn; }
  *(v8f*)(qbf + (size_t)j * CP + half * 8)      = o0;
  *(v8f*)(qbf + (size_t)j * CP + 16 + half * 8) = o1;
}

// ---------------- separable spatial gaussian ----------------
__global__ void k_sp_h(const float* __restrict__ q, const float* __restrict__ g,
                       float* __restrict__ tmp) {
  int idx = blockIdx.x * blockDim.x + threadIdx.x;
  if (idx >= NC * NPIX) return;
  int c = idx / NPIX, rem = idx % NPIX;
  int y = rem / WW, x = rem % WW;
  float s = 0.f;
  for (int t = 0; t < KS; ++t) {
    int xs = x + t - RAD;
    if (xs >= 0 && xs < WW) s += g[t] * q[c * NPIX + y * WW + xs];
  }
  tmp[idx] = s;
}

__global__ void k_sp_v(const float* __restrict__ tmp, const float* __restrict__ g,
                       float* __restrict__ qsf) {
  int idx = blockIdx.x * blockDim.x + threadIdx.x;
  if (idx >= NC * NPIX) return;
  int c = idx / NPIX, rem = idx % NPIX;
  int y = rem / WW, x = rem % WW;
  float s = 0.f;
  for (int t = 0; t < KS; ++t) {
    int ys = y + t - RAD;
    if (ys >= 0 && ys < HH) s += g[t] * tmp[c * NPIX + ys * WW + x];
  }
  qsf[idx] = s;
}

// ---------------- combine + softmax ----------------
__global__ void k_combine(const float* __restrict__ U, const float* __restrict__ qbf,
                          const float* __restrict__ qsf, float* __restrict__ q,
                          float* __restrict__ out) {
  int p = blockIdx.x * blockDim.x + threadIdx.x;
  if (p >= NPIX) return;
  float h[NC]; float mx = -1e30f;
#pragma unroll
  for (int c = 0; c < NC; ++c) {
    float v = U[c * NPIX + p] + COMPAT_BF * qbf[(size_t)p * CP + c]
            + COMPAT_SP * qsf[c * NPIX + p];
    h[c] = v; mx = fmaxf(mx, v);
  }
  float s = 0.f;
#pragma unroll
  for (int c = 0; c < NC; ++c) { h[c] = __expf(h[c] - mx); s += h[c]; }
  float inv = __frcp_rn(s);
#pragma unroll
  for (int c = 0; c < NC; ++c) {
    float v = h[c] * inv;
    q[c * NPIX + p] = v;
    if (out) out[c * NPIX + p] = v;
  }
}

// ---------------- launch ----------------
extern "C" void kernel_launch(void* const* d_in, const int* in_sizes, int n_in,
                              void* d_out, int out_size, void* d_ws, size_t ws_size,
                              hipStream_t stream) {
  const float* unary = (const float*)d_in[0];
  const float* ref   = (const float*)d_in[1];
  const float* kstd  = (const float*)d_in[2];

  char* ws = (char*)d_ws;
  size_t off = 0;
  auto alloc = [&](size_t bytes) -> void* {
    void* p = ws + off;
    off = (off + bytes + 255) & ~(size_t)255;
    return p;
  };
  float*        F    = (float*)alloc((size_t)NPIX * NF * 4);
  float*        sq   = (float*)alloc((size_t)NPIX * 4);
  float*        U    = (float*)alloc((size_t)NC * NPIX * 4);
  float*        q    = (float*)alloc((size_t)NC * NPIX * 4);
  float*        nrm  = (float*)alloc((size_t)NPIX * 4);
  float*        g1d  = (float*)alloc((size_t)KS * 4);
  unsigned int* qn   = (unsigned int*)alloc((size_t)NPIX * CP * 2);
  float*        qbf  = (float*)alloc((size_t)NPIX * CP * 4);
  float*        tmpS = (float*)alloc((size_t)NC * NPIX * 4);
  float*        qsf  = (float*)alloc((size_t)NC * NPIX * 4);

  const int PB = NPIX / 256;            // 36
  const int CB = (NC * NPIX + 255) / 256;

  k_init_feat<<<PB, 256, 0, stream>>>(ref, kstd, F, sq);
  k_init_uq<<<PB, 256, 0, stream>>>(unary, U, q);
  k_g1d<<<1, 128, 0, stream>>>(g1d);
  k_norm<<<PB, 256, 0, stream>>>(F, sq, nrm);

  for (int it = 0; it < 5; ++it) {
    k_prep<<<NPIX / 2 / 256, 256, 0, stream>>>(q, nrm, qn);
    k_bilateral<<<NPIX / 128, 256, 0, stream>>>(F, sq, qn, nrm, qbf);
    k_sp_h<<<CB, 256, 0, stream>>>(q, g1d, tmpS);
    k_sp_v<<<CB, 256, 0, stream>>>(tmpS, g1d, qsf);
    float* outp = (it == 4) ? (float*)d_out : nullptr;
    k_combine<<<PB, 256, 0, stream>>>(U, qbf, qsf, q, outp);
  }
}
